// GNODE_14130442403994
// MI455X (gfx1250) — matrix operand order repaired and checked by reading, not compiled, not run on hardware
//
#include <hip/hip_runtime.h>

// ------------------------------------------------------------------
// types
// ------------------------------------------------------------------
typedef __bf16 bf16;
typedef __attribute__((ext_vector_type(16))) __bf16 v16bf;
typedef __attribute__((ext_vector_type(8)))  __bf16 v8bf;
typedef __attribute__((ext_vector_type(2)))  __bf16 v2bf;
typedef __attribute__((ext_vector_type(8)))  float  v8f;

// ------------------------------------------------------------------
// problem constants
// ------------------------------------------------------------------
#define N_NODES   100000
#define N_EDGES   99999
#define KP        224          // IN_DIM 196 padded to 7*32 (col 196 = constant 1.0 bias column)
#define NTL3      13           // live layer-3 feature tiles (tile 13 = pure zero padding, skipped)
#define DT_F      0.02f
#define RKH       0.005f       // (HORIZON*DT)/NSTEPS
#define NTILES    6250         // N_NODES / 16  (exact)
#define WPB       4            // waves per block
#define THREADS   128
#define NBLOCKS   ((NTILES + WPB - 1) / WPB)

// ------------------------------------------------------------------
// workspace (d_ws) layout, bytes
// ------------------------------------------------------------------
#define WS_HE        ((size_t)0)
#define HE_BYTES     ((size_t)N_NODES * 64 * 2)        // 12,800,000
#define WS_STATIC    (HE_BYTES)
#define STATIC_BYTES 30016
#define WS_ROUND0    (WS_STATIC + STATIC_BYTES)
#define ROUND_BYTES  95872
#define WS_ROUND1    (WS_ROUND0 + ROUND_BYTES)

// offsets inside static blob (bf16 weights transposed, f32 biases)
#define SB_ENCW1 0        // [64][32]  bf16   (enc_node W1^T, K pad 6->32, row6 = b1 folded)
#define SB_ENCW2 4096     // [64][64]  bf16
#define SB_DECW1 12288    // [64][64]  bf16
#define SB_DECW2 20480    // [64][64]  bf16
#define SB_ENCB1 28672    // f32[64]  (kept, unused by main)
#define SB_ENCB2 28928
#define SB_DECB1 29184
#define SB_DECB2 29440
#define SB_DECW3 29696    // f32[64]
#define SB_DECB3 29952    // f32[1] (padded 16B)
#define SB_CONTENT 29968  // multiple of 16

// offsets inside a per-round blob
#define RB_WD1T 0         // [64][224]  bf16  (deriv W1^T, row196 = bd1 folded)
#define RB_WD2T 28672     // [64][64]   bf16
#define RB_WD3T 36864     // [224][64]  bf16  (deriv W3^T, N pad 196->224 zeros)
#define RB_WPT  65536     // [64][224]  bf16  (proj W^T, row196 = bp folded)
#define RB_BD1  94208     // f32[64]  (kept, unused by main)
#define RB_BD2  94464     // f32[64]
#define RB_BD3  94720     // f32[224] (padded zeros beyond 195)
#define RB_BP   95616     // f32[64]  (kept, unused by main)

// ------------------------------------------------------------------
// dynamic LDS layout (per block)
// ------------------------------------------------------------------
#define LDS_ROUND   STATIC_BYTES                 // 30016
#define LDS_WAVE0   (LDS_ROUND + ROUND_BYTES)    // 125888
#define WV_X0F  0        // [16][224] f32  (round-start staging only)
#define WV_XTH  14336    // [16][224] bf16
#define WV_YA   21504    // [16][64]  bf16
#define WV_YB   23552    // [16][64]  bf16
#define WV_HN   25600    // [16][64]  f32
#define WAVE_BYTES 29696
#define LDS_TOTAL (LDS_WAVE0 + WPB * WAVE_BYTES) // 244672 < 320KB

// ------------------------------------------------------------------
// parameter pointer bundle (resolved on host)
// ------------------------------------------------------------------
struct P {
    const float *nodes, *edges, *g;
    const float *encnW1, *encnB1, *encnW2, *encnB2;
    const float *enceW1, *enceB1, *enceW2, *enceB2;
    const float *d0W1, *d0B1, *d0W2, *d0B2, *d0W3, *d0B3;
    const float *d1W1, *d1B1, *d1W2, *d1B2, *d1W3, *d1B3;
    const float *p0W, *p0B, *p1W, *p1B;
    const float *dnW1, *dnB1, *dnW2, *dnB2, *dnW3, *dnB3;
};

// ------------------------------------------------------------------
// device helpers
// ------------------------------------------------------------------
__device__ __forceinline__ v8f vzero() {
    v8f z = {0.f, 0.f, 0.f, 0.f, 0.f, 0.f, 0.f, 0.f};
    return z;
}

__device__ __forceinline__ float relu1(float y) {
    return __builtin_amdgcn_fmed3f(y, 0.f, __builtin_inff());
}

// pack 8 f32 -> 8 bf16 (4x v_cvt_pk_bf16_f32) and store as one b128
__device__ __forceinline__ void store_pk(bf16* dst, const float* y) {
    v8bf pk;
#pragma unroll
    for (int v = 0; v < 8; ++v) pk[v] = (bf16)y[v];
    *(v8bf*)dst = pk;
}

// same but with relu done in the packed bf16 domain:
// cvt_pk canonicalizes, then one v_pk_max_num_bf16 per pair (no canonicalize-max pairs)
__device__ __forceinline__ void store_pk_relu(bf16* dst, const float* y) {
    union { unsigned u[4]; v8bf h; } pk;
#pragma unroll
    for (int h2 = 0; h2 < 4; ++h2) {
        union { v2bf h; unsigned u; } t;
        t.h[0] = (bf16)y[2 * h2];
        t.h[1] = (bf16)y[2 * h2 + 1];
        unsigned r;
        asm("v_pk_max_num_bf16 %0, %1, 0" : "=v"(r) : "v"(t.u));
        pk.u[h2] = r;
    }
    *(v8bf*)dst = pk.h;
}

// A fragment (16x32 bf16): weights W^T stored [outF][kpitch]; mt selects 16-feature row tile.
__device__ __forceinline__ v16bf loadAw(const bf16* wt, int kpitch, int kc, int mt, int lane) {
    const bf16* p = wt + (mt * 16 + (lane & 15)) * kpitch + kc * 32 + ((lane >> 4) << 4);
    v8bf lo = *(const v8bf*)p;
    v8bf hi = *(const v8bf*)(p + 8);
    v16bf a;
#pragma unroll
    for (int i = 0; i < 8; ++i) { a[i] = lo[i]; a[i + 8] = hi[i]; }
    return a;
}

// B fragment (32x16 bf16): activations stored row-major [node][pitch]; one 32B read.
__device__ __forceinline__ v16bf loadBx(const bf16* s, int pitch, int kc, int lane) {
    return *(const v16bf*)(s + (lane & 15) * pitch + kc * 32 + ((lane >> 4) << 4));
}

#define WMMA_BF16(a, b, c) \
    __builtin_amdgcn_wmma_f32_16x16x32_bf16(false, (a), false, (b), (short)0, (c), false, false)

// transposed-orientation GEMM: D[feat, node] = W^T(A) * act^T(B).
// Each lane owns one node and 8 consecutive features per tile -> packed stores.
template <int KC, int MT, bool RELU, bool ACCUM, typename TD, bool BIAS = true>
__device__ __forceinline__ void gemm_tileT(const bf16* __restrict__ src, int spitch,
                                           const bf16* __restrict__ wt, int wpitch,
                                           const float* __restrict__ bias,
                                           TD* __restrict__ dst, int dpitch, int lane) {
    v8f c[MT];
#pragma unroll
    for (int mt = 0; mt < MT; ++mt) c[mt] = vzero();
#pragma unroll
    for (int kc = 0; kc < KC; ++kc) {
        v16bf b = loadBx(src, spitch, kc, lane);
#pragma unroll
        for (int mt = 0; mt < MT; ++mt) {
            v16bf a = loadAw(wt, wpitch, kc, mt, lane);
            c[mt] = WMMA_BF16(a, b, c[mt]);
        }
    }
    const int node = lane & 15, fb = (lane >> 4) * 8;
#pragma unroll
    for (int mt = 0; mt < MT; ++mt) {
        const int fbase = mt * 16 + fb;
        const int idx = node * dpitch + fbase;
        float y[8];
#pragma unroll
        for (int v = 0; v < 8; ++v) {
            y[v] = c[mt][v];
            if (BIAS) y[v] += bias[fbase + v];
        }
        if (ACCUM) {
#pragma unroll
            for (int v = 0; v < 8; ++v) {
                if (RELU) y[v] = relu1(y[v]);
                y[v] += (float)dst[idx + v];
                dst[idx + v] = (TD)y[v];
            }
        } else if (sizeof(TD) == 2) {
            if (RELU) store_pk_relu((bf16*)dst + idx, y);
            else      store_pk((bf16*)dst + idx, y);
        } else {
#pragma unroll
            for (int v = 0; v < 8; ++v) {
                if (RELU) y[v] = relu1(y[v]);
                dst[idx + v] = (TD)y[v];
            }
        }
    }
}

// one f(x) evaluation + RK4 combine; x0/acc in registers, layer-2/3 weight
// A-fragments pre-hoisted into registers (round-invariant).
__device__ __forceinline__ void rk_evalT(float acc[NTL3][8], float x0r[NTL3][8],
                                         const v16bf (&a3)[2][NTL3], const v16bf (&a2)[2][4],
                                         const bf16* wd1t, const float* bd2, const float* bd3,
                                         bf16* xth, bf16* ybA, bf16* ybB,
                                         float wa, float wx, bool finale, int lane) {
    // layer1: y1 = relu(x @ Wd1)  (bias folded via col196==1)
    gemm_tileT<7, 4, true, false, bf16, false>(xth, KP, wd1t, KP, nullptr, ybA, 64, lane);
    const int node = lane & 15, fb = (lane >> 4) * 8;
    // layer2: y2 = relu(y1 @ Wd2 + bd2)
    {
        v16bf b0 = loadBx(ybA, 64, 0, lane);
        v16bf b1 = loadBx(ybA, 64, 1, lane);
#pragma unroll
        for (int mt = 0; mt < 4; ++mt) {
            v8f c = vzero();
            c = WMMA_BF16(a2[0][mt], b0, c);
            c = WMMA_BF16(a2[1][mt], b1, c);
            const int fbase = mt * 16 + fb;
            float y[8];
#pragma unroll
            for (int v = 0; v < 8; ++v) y[v] = c[v] + bd2[fbase + v];
            store_pk_relu(ybB + node * 64 + fbase, y);
        }
    }
    // layer3 + RK combine (tile 13 is all-zero padding; skipped)
    v16bf b0 = loadBx(ybB, 64, 0, lane);
    v16bf b1 = loadBx(ybB, 64, 1, lane);
#pragma unroll
    for (int nt = 0; nt < NTL3; ++nt) {
        v8f c = vzero();
        c = WMMA_BF16(a3[0][nt], b0, c);
        c = WMMA_BF16(a3[1][nt], b1, c);
        const int fbase = nt * 16 + fb;
        float y[8];
#pragma unroll
        for (int v = 0; v < 8; ++v) {
            float k = c[v] + bd3[fbase + v];
            acc[nt][v] += wa * k;
            if (!finale) {
                y[v] = x0r[nt][v] + wx * k;
            } else {
                y[v] = acc[nt][v];
                x0r[nt][v] = y[v];
            }
        }
        store_pk(xth + node * KP + fbase, y);
    }
}

// block-cooperative global -> LDS copy using the CDNA5 async-to-LDS path.
__device__ __forceinline__ void copyBlobAsync(char* ldsDst, const char* src, int bytes) {
    unsigned ldsBase = (unsigned)(unsigned long long)ldsDst;  // addr[31:0] = LDS offset
    unsigned long long gBase = (unsigned long long)src;
    int n = bytes >> 4;
    for (int i = threadIdx.x; i < n; i += blockDim.x) {
        unsigned loff = ldsBase + ((unsigned)i << 4);
        unsigned long long ga = gBase + ((unsigned long long)i << 4);
        asm volatile("global_load_async_to_lds_b128 %0, %1, off"
                     :: "v"(loff), "v"(ga) : "memory");
    }
    asm volatile("s_wait_asynccnt 0x0" ::: "memory");
}

// ------------------------------------------------------------------
// kernel 1: weight transpose/pad/convert to bf16 in ws
// ------------------------------------------------------------------
__global__ void __launch_bounds__(256) prep_weights(P p, char* ws) {
    int tid = blockIdx.x * blockDim.x + threadIdx.x;
    int str = gridDim.x * blockDim.x;
    bf16* o;
    float* f;
    o = (bf16*)(ws + WS_STATIC + SB_ENCW1);
    for (int i = tid; i < 64 * 32; i += str) {
        int n = i >> 5, k = i & 31;
        float v = (k < 6) ? p.encnW1[k * 64 + n] : (k == 6 ? p.encnB1[n] : 0.f);
        o[i] = (bf16)v;
    }
    o = (bf16*)(ws + WS_STATIC + SB_ENCW2);
    for (int i = tid; i < 64 * 64; i += str) { int n = i >> 6, k = i & 63; o[i] = (bf16)p.encnW2[k * 64 + n]; }
    o = (bf16*)(ws + WS_STATIC + SB_DECW1);
    for (int i = tid; i < 64 * 64; i += str) { int n = i >> 6, k = i & 63; o[i] = (bf16)p.dnW1[k * 64 + n]; }
    o = (bf16*)(ws + WS_STATIC + SB_DECW2);
    for (int i = tid; i < 64 * 64; i += str) { int n = i >> 6, k = i & 63; o[i] = (bf16)p.dnW2[k * 64 + n]; }
    f = (float*)(ws + WS_STATIC + SB_ENCB1);
    for (int i = tid; i < 64; i += str) f[i] = p.encnB1[i];
    f = (float*)(ws + WS_STATIC + SB_ENCB2);
    for (int i = tid; i < 64; i += str) f[i] = p.encnB2[i];
    f = (float*)(ws + WS_STATIC + SB_DECB1);
    for (int i = tid; i < 64; i += str) f[i] = p.dnB1[i];
    f = (float*)(ws + WS_STATIC + SB_DECB2);
    for (int i = tid; i < 64; i += str) f[i] = p.dnB2[i];
    f = (float*)(ws + WS_STATIC + SB_DECW3);
    for (int i = tid; i < 64; i += str) f[i] = p.dnW3[i];   // W3 is [64][1]
    f = (float*)(ws + WS_STATIC + SB_DECB3);
    if (tid == 0) f[0] = p.dnB3[0];
    for (int r = 0; r < 2; ++r) {
        const float *W1 = r ? p.d1W1 : p.d0W1, *B1 = r ? p.d1B1 : p.d0B1;
        const float *W2 = r ? p.d1W2 : p.d0W2, *B2 = r ? p.d1B2 : p.d0B2;
        const float *W3 = r ? p.d1W3 : p.d0W3, *B3 = r ? p.d1B3 : p.d0B3;
        const float *Wp = r ? p.p1W : p.p0W,   *Bp = r ? p.p1B : p.p0B;
        char* rb = ws + (r ? WS_ROUND1 : WS_ROUND0);
        o = (bf16*)(rb + RB_WD1T);
        for (int i = tid; i < 64 * 224; i += str) {
            int n = i / 224, k = i % 224;
            float v = (k < 196) ? W1[k * 64 + n] : (k == 196 ? B1[n] : 0.f);
            o[i] = (bf16)v;
        }
        o = (bf16*)(rb + RB_WD2T);
        for (int i = tid; i < 64 * 64; i += str) { int n = i >> 6, k = i & 63; o[i] = (bf16)W2[k * 64 + n]; }
        o = (bf16*)(rb + RB_WD3T);
        for (int i = tid; i < 224 * 64; i += str) { int n = i >> 6, k = i & 63; o[i] = (bf16)(n < 196 ? W3[k * 196 + n] : 0.f); }
        o = (bf16*)(rb + RB_WPT);
        for (int i = tid; i < 64 * 224; i += str) {
            int n = i / 224, k = i % 224;
            float v = (k < 196) ? Wp[k * 64 + n] : (k == 196 ? Bp[n] : 0.f);
            o[i] = (bf16)v;
        }
        f = (float*)(rb + RB_BD1);
        for (int i = tid; i < 64; i += str) f[i] = B1[i];
        f = (float*)(rb + RB_BD2);
        for (int i = tid; i < 64; i += str) f[i] = B2[i];
        f = (float*)(rb + RB_BD3);
        for (int i = tid; i < 224; i += str) f[i] = (i < 196) ? B3[i] : 0.f;
        f = (float*)(rb + RB_BP);
        for (int i = tid; i < 64; i += str) f[i] = Bp[i];
    }
}

// ------------------------------------------------------------------
// kernel 2: edge encoder  h_e0 = MLP_1->64->64(edges)  (bf16 rows in ws)
// ------------------------------------------------------------------
__global__ void __launch_bounds__(256) edge_enc(const float* __restrict__ edges,
                                                const float* __restrict__ W1,
                                                const float* __restrict__ B1,
                                                const float* __restrict__ W2,
                                                const float* __restrict__ B2,
                                                bf16* __restrict__ he) {
    __shared__ float sW2[64 * 64];
    __shared__ float sW1[64], sB1[64], sB2[64];
    for (int i = threadIdx.x; i < 64 * 64; i += 256) sW2[i] = W2[i];
    if (threadIdx.x < 64) {
        sW1[threadIdx.x] = W1[threadIdx.x];
        sB1[threadIdx.x] = B1[threadIdx.x];
        sB2[threadIdx.x] = B2[threadIdx.x];
    }
    __syncthreads();
    int e = blockIdx.x * 256 + threadIdx.x;
    if (e >= N_EDGES) return;
    float v = edges[e];
    float h1[64];
#pragma unroll
    for (int k = 0; k < 64; ++k) h1[k] = relu1(v * sW1[k] + sB1[k]);
    bf16* o = he + (size_t)e * 64;
    for (int n = 0; n < 64; ++n) {
        float a = sB2[n];
#pragma unroll
        for (int k = 0; k < 64; ++k) a += h1[k] * sW2[k * 64 + n];
        o[n] = (bf16)a;
    }
}

// ------------------------------------------------------------------
// kernel 3: main fused GNODE (encoder + 2x message-passing RK4 + decoder)
// ------------------------------------------------------------------
__global__ void __launch_bounds__(THREADS) gnode_main(const float* __restrict__ nodes,
                                                      const float* __restrict__ g4,
                                                      const char* __restrict__ ws,
                                                      float* __restrict__ out) {
    extern __shared__ __align__(128) char smem[];
    const bf16* he = (const bf16*)(ws + WS_HE);

    copyBlobAsync(smem, ws + WS_STATIC, SB_CONTENT);

    const int w = threadIdx.x >> 5, lane = threadIdx.x & 31;
    const int gw = blockIdx.x * WPB + w;
    const bool active = gw < NTILES;
    const int tile = active ? gw : 0;          // idle waves redo tile 0, writes masked
    const int m = lane & 15;

    if (lane < 16) {
        int j = tile * 16 + m;
        __builtin_prefetch(he + (size_t)j * 64, 0, 1);
    }
    __syncthreads();

    char* wb = smem + LDS_WAVE0 + w * WAVE_BYTES;
    float* x0f = (float*)(wb + WV_X0F);
    bf16*  xth = (bf16*)(wb + WV_XTH);
    bf16*  ybA = (bf16*)(wb + WV_YA);
    bf16*  ybB = (bf16*)(wb + WV_YB);
    float* hn  = (float*)(wb + WV_HN);

    const bf16*  encW1t = (const bf16*)(smem + SB_ENCW1);
    const bf16*  encW2t = (const bf16*)(smem + SB_ENCW2);
    const bf16*  decW1t = (const bf16*)(smem + SB_DECW1);
    const bf16*  decW2t = (const bf16*)(smem + SB_DECW2);
    const float* encB2  = (const float*)(smem + SB_ENCB2);
    const float* decB1  = (const float*)(smem + SB_DECB1);
    const float* decB2  = (const float*)(smem + SB_DECB2);
    const float* decW3  = (const float*)(smem + SB_DECW3);
    const float* decB3  = (const float*)(smem + SB_DECB3);

    const bf16*  wd1t = (const bf16*)(smem + LDS_ROUND + RB_WD1T);
    const bf16*  wd2t = (const bf16*)(smem + LDS_ROUND + RB_WD2T);
    const bf16*  wd3t = (const bf16*)(smem + LDS_ROUND + RB_WD3T);
    const bf16*  wpt  = (const bf16*)(smem + LDS_ROUND + RB_WPT);
    const float* bd2  = (const float*)(smem + LDS_ROUND + RB_BD2);
    const float* bd3  = (const float*)(smem + LDS_ROUND + RB_BD3);

    // ---- node encoder: h_n = (relu(nodes@W1+b1))@W2+b2 ; b1 folded via col6==1 ----
    if (lane < 16) {
        int j = tile * 16 + m;
#pragma unroll
        for (int c = 0; c < 6; ++c) xth[m * KP + c] = (bf16)nodes[(size_t)j * 6 + c];
        xth[m * KP + 6] = (bf16)1.f;
#pragma unroll
        for (int c = 7; c < 32; ++c) xth[m * KP + c] = (bf16)0.f;
    }
    gemm_tileT<1, 4, true,  false, bf16, false>(xth, KP, encW1t, 32, nullptr, ybA, 64, lane);
    gemm_tileT<2, 4, false, false, float>(ybA, 64, encW2t, 64, encB2, hn, 64, lane);

    const float g1 = g4[1], g2 = g4[2], g3 = g4[3];

    // ---- message-passing rounds ----
    for (int r = 0; r < 2; ++r) {
        __syncthreads();
        copyBlobAsync(smem + LDS_ROUND, ws + (r ? WS_ROUND1 : WS_ROUND0), ROUND_BYTES);
        __syncthreads();

        // hoist round-invariant layer-2/3 weight A-fragments into registers
        v16bf a3[2][NTL3], a2[2][4];
#pragma unroll
        for (int kc = 0; kc < 2; ++kc) {
#pragma unroll
            for (int nt = 0; nt < NTL3; ++nt) a3[kc][nt] = loadAw(wd3t, 64, kc, nt, lane);
#pragma unroll
            for (int mt = 0; mt < 4; ++mt)   a2[kc][mt] = loadAw(wd2t, 64, kc, mt, lane);
        }

        const float scale = r ? 2.f : 1.f;         // h_e doubles each round
        const float g0 = g4[0] + (float)r;         // g0 + r*HORIZON

        // build x0 = [h_n | scale*he[j] | scale*he[j-1] | g | 1.0 | 0-pad]
        {
            int jj = tile * 16 + m;
            int half = lane >> 4;
            for (int t = 0; t < 112; ++t) {
                int c = half + 2 * t;
                float val;
                if (c < 64)        val = hn[m * 64 + c];
                else if (c < 128)  val = (jj < N_NODES - 1) ? scale * (float)he[(size_t)jj * 64 + (c - 64)] : 0.f;
                else if (c < 192)  val = (jj > 0) ? scale * (float)he[(size_t)(jj - 1) * 64 + (c - 128)] : 0.f;
                else if (c == 192) val = g0;
                else if (c == 193) val = g1;
                else if (c == 194) val = g2;
                else if (c == 195) val = g3;
                else if (c == 196) val = 1.f;      // bias column (self-preserving under RK4)
                else               val = 0.f;
                x0f[m * KP + c] = val;
                xth[m * KP + c] = (bf16)val;
            }
        }
        // x0 and acc in registers: lane owns node (lane&15), 8 consecutive feats per tile
        float acc[NTL3][8], x0r[NTL3][8];
        {
            const int node = lane & 15, fb = (lane >> 4) * 8;
#pragma unroll
            for (int nt = 0; nt < NTL3; ++nt)
#pragma unroll
                for (int v = 0; v < 8; ++v) {
                    float t = x0f[node * KP + nt * 16 + fb + v];
                    x0r[nt][v] = t;
                    acc[nt][v] = t;
                }
        }
        // RK4, 4 steps x 4 evals
        for (int s = 0; s < 4; ++s) {
            rk_evalT(acc, x0r, a3, a2, wd1t, bd2, bd3, xth, ybA, ybB,
                     RKH * (1.f / 6.f), RKH * 0.5f, false, lane);
            rk_evalT(acc, x0r, a3, a2, wd1t, bd2, bd3, xth, ybA, ybB,
                     RKH * (1.f / 3.f), RKH * 0.5f, false, lane);
            rk_evalT(acc, x0r, a3, a2, wd1t, bd2, bd3, xth, ybA, ybB,
                     RKH * (1.f / 3.f), RKH, false, lane);
            rk_evalT(acc, x0r, a3, a2, wd1t, bd2, bd3, xth, ybA, ybB,
                     RKH * (1.f / 6.f), 0.f, true, lane);
        }
        // h_n += proj(xT)   (bp folded via col196==1)
        gemm_tileT<7, 4, false, true, float, false>(xth, KP, wpt, KP, nullptr, hn, 64, lane);
    }

    // ---- decoder: relu(relu(hn@W1+b1)@W2+b2) . w3 + b3 ----
    if (lane < 16) {
#pragma unroll
        for (int c = 0; c < 64; ++c) xth[m * KP + c] = (bf16)hn[m * 64 + c];
    }
    gemm_tileT<2, 4, true, false, bf16>(xth, KP, decW1t, 64, decB1, ybA, 64, lane);
    gemm_tileT<2, 4, true, false, bf16>(ybA, 64, decW2t, 64, decB2, ybB, 64, lane);

    if (active && lane < 16) {
        float s = decB3[0];
#pragma unroll
        for (int c = 0; c < 64; ++c) s += (float)ybB[m * 64 + c] * decW3[c];
        int j = tile * 16 + m;
        const float* nd = nodes + (size_t)j * 6;
        float next_vel = nd[5] + s * DT_F;
        float next_pos = nd[0] + next_vel * DT_F;
        float* o = out + (size_t)j * 7;
        o[0] = next_pos; o[1] = nd[2]; o[2] = nd[3]; o[3] = nd[4];
        o[4] = nd[5];    o[5] = next_vel; o[6] = s;
    }
}

// ------------------------------------------------------------------
// kernel 4: next_edges = diff(next_pos)
// ------------------------------------------------------------------
__global__ void __launch_bounds__(256) diff_edges(float* __restrict__ out) {
    int j = blockIdx.x * 256 + threadIdx.x;
    if (j < N_EDGES)
        out[(size_t)N_NODES * 7 + j] = out[(size_t)(j + 1) * 7] - out[(size_t)j * 7];
}

// ------------------------------------------------------------------
// host entry
// ------------------------------------------------------------------
extern "C" void kernel_launch(void* const* d_in, const int* in_sizes, int n_in,
                              void* d_out, int out_size, void* d_ws, size_t ws_size,
                              hipStream_t stream) {
    (void)out_size; (void)ws_size;
    static const int IDX_INS[30] = {5,6,7,8, 9,10,11,12, 13,14,15,16,17,18,
                                    19,20,21,22,23,24, 25,26,27,28, 29,30,31,32,33,34};
    static const int IDX_SRT[30] = {33,34,35,36, 29,30,31,32, 17,18,19,20,21,22,
                                    23,24,25,26,27,28, 37,38,39,40, 11,12,13,14,15,16};
    const bool insertion = (n_in >= 35 && in_sizes[5] == 384);  // enc_node W1 = 6*64
    const int* I = insertion ? IDX_INS : IDX_SRT;

    P p;
    p.nodes = (const float*)d_in[0];
    p.edges = (const float*)d_in[1];
    p.g     = (const float*)d_in[2];
    const float* q[30];
    for (int i = 0; i < 30; ++i) q[i] = (const float*)d_in[I[i]];
    p.encnW1 = q[0];  p.encnB1 = q[1];  p.encnW2 = q[2];  p.encnB2 = q[3];
    p.enceW1 = q[4];  p.enceB1 = q[5];  p.enceW2 = q[6];  p.enceB2 = q[7];
    p.d0W1 = q[8];  p.d0B1 = q[9];  p.d0W2 = q[10]; p.d0B2 = q[11]; p.d0W3 = q[12]; p.d0B3 = q[13];
    p.d1W1 = q[14]; p.d1B1 = q[15]; p.d1W2 = q[16]; p.d1B2 = q[17]; p.d1W3 = q[18]; p.d1B3 = q[19];
    p.p0W = q[20]; p.p0B = q[21]; p.p1W = q[22]; p.p1B = q[23];
    p.dnW1 = q[24]; p.dnB1 = q[25]; p.dnW2 = q[26]; p.dnB2 = q[27]; p.dnW3 = q[28]; p.dnB3 = q[29];

    char* ws = (char*)d_ws;
    (void)hipFuncSetAttribute(reinterpret_cast<const void*>(gnode_main),
                              hipFuncAttributeMaxDynamicSharedMemorySize, (int)LDS_TOTAL);

    prep_weights<<<64, 256, 0, stream>>>(p, ws);
    edge_enc<<<(N_EDGES + 255) / 256, 256, 0, stream>>>(p.edges, p.enceW1, p.enceB1,
                                                        p.enceW2, p.enceB2,
                                                        (bf16*)(ws + WS_HE));
    gnode_main<<<NBLOCKS, THREADS, LDS_TOTAL, stream>>>(p.nodes, p.g, ws, (float*)d_out);
    diff_edges<<<(N_EDGES + 255) / 256, 256, 0, stream>>>((float*)d_out);
}
